// CrossAttention_59871844106349
// MI455X (gfx1250) — matrix operand order repaired and checked by reading, not compiled, not run on hardware
//
#include <hip/hip_runtime.h>
#include <hip/hip_bf16.h>

typedef __attribute__((ext_vector_type(16))) __bf16 v16bf;
typedef __attribute__((ext_vector_type(8)))  float  v8f;
typedef __attribute__((ext_vector_type(4)))  float  f32x4;
typedef __attribute__((ext_vector_type(4)))  unsigned int u32x4;

#define B_    4
#define C_    256
#define CQK_  32
#define N_    4096
#define NT16_ (N_/16)
#define LOG2E_ 1.4426950408889634f

union V16U { v16bf v; u32x4 q[2]; };          // 32-byte A/B operand
union PK8  { u32x4 q; __bf16 h[8]; };          // 8x bf16 pack for 16B stores

static __device__ __forceinline__ __bf16 f2bf(float f) { return (__bf16)f; }

static __device__ __forceinline__ v8f wmma_bf16(v16bf a, v16bf b, v8f c) {
  return __builtin_amdgcn_wmma_f32_16x16x32_bf16(false, a, false, b, (short)0, c,
                                                 false, false);
}

// Wave-local LDS fence: same-wave DS ops complete in order; we only need to
// stop the compiler from reordering and drain DScnt before dependent reads.
static __device__ __forceinline__ void lds_fence_wave() {
  asm volatile("s_wait_dscnt 0x0" ::: "memory");
}

// ---------------------------------------------------------------------------
// Phase 1: q/k/v projections.  One wave = one 16-column tile of x.
//   Qh[b][n][o] (bf16, PRE-SCALED by log2(e)), Kh[b][n][o], Vt[b][c][n].
// ---------------------------------------------------------------------------
__global__ __launch_bounds__(128) void proj_kernel(
    const float* __restrict__ x,
    const float* __restrict__ wq, const float* __restrict__ bq,
    const float* __restrict__ wk, const float* __restrict__ bk,
    const float* __restrict__ wv, const float* __restrict__ bv,
    __bf16* __restrict__ Qh, __bf16* __restrict__ Kh, __bf16* __restrict__ Vt)
{
  // x^T tile per wave: [n 0..15][c 0..255], padded row (264 bf16 = 33*16B).
  __shared__ __attribute__((aligned(16))) __bf16 Xs[4][16][264];

  const int w    = threadIdx.x >> 5;
  const int lane = threadIdx.x & 31;
  const int hf   = lane >> 4;     // lane half (0/1)
  const int r    = lane & 15;
  const int gt   = blockIdx.x * 4 + w;
  const int b    = gt / NT16_;
  const int n0   = (gt % NT16_) * 16;

  // ---- load x[b][:, n0:n0+16] and transpose into LDS as bf16 ----
  #pragma unroll
  for (int it = 0; it < 8; ++it) {
    const int c = lane + it * 32;
    const float* xp = x + ((size_t)(b * C_ + c)) * N_ + n0;
    f32x4 f0 = *(const f32x4*)(xp + 0);
    f32x4 f1 = *(const f32x4*)(xp + 4);
    f32x4 f2 = *(const f32x4*)(xp + 8);
    f32x4 f3 = *(const f32x4*)(xp + 12);
    float fv[16] = {f0.x,f0.y,f0.z,f0.w, f1.x,f1.y,f1.z,f1.w,
                    f2.x,f2.y,f2.z,f2.w, f3.x,f3.y,f3.z,f3.w};
    #pragma unroll
    for (int j = 0; j < 16; ++j) Xs[w][j][c] = f2bf(fv[j]);
  }
  lds_fence_wave();

  // ---- q and k projections: 2 output tiles each, 8 K-chunks of 32 ----
  #pragma unroll
  for (int pqk = 0; pqk < 2; ++pqk) {
    const float* W    = pqk ? wk : wq;
    const float* bias = pqk ? bk : bq;
    __bf16*      Outp = pqk ? Kh : Qh;
    const float  sc   = pqk ? 1.0f : LOG2E_;   // fold log2(e) into Q
    #pragma unroll
    for (int ot = 0; ot < 2; ++ot) {
      const int orow = ot * 16 + r;            // this lane's A-matrix row
      v8f acc = {};
      #pragma unroll
      for (int kc = 0; kc < 8; ++kc) {
        // A tile (16x32 of W): lane row=orow, K = (j/8)*16 + hf*8 + j%8
        V16U a;
        const float* wr = W + orow * C_ + kc * 32;
        #pragma unroll
        for (int j = 0; j < 8; ++j) {
          a.v[j]     = f2bf(wr[hf * 8 + j]);
          a.v[8 + j] = f2bf(wr[16 + hf * 8 + j]);
        }
        // B tile (32x16 of x^T): lane col=r, K = (j/8)*16 + hf*8 + j%8
        V16U bm;
        const u32x4* bp = (const u32x4*)&Xs[w][r][kc * 32 + hf * 8];
        bm.q[0] = bp[0]; bm.q[1] = bp[2];
        acc = wmma_bf16(a.v, bm.v, acc);
      }
      // D layout: n = r, o = ot*16 + hf*8 + i -> 8 contiguous bf16 = one b128
      const int obase = ot * 16 + hf * 8;
      PK8 pk;
      #pragma unroll
      for (int i = 0; i < 8; ++i)
        pk.h[i] = f2bf((acc[i] + bias[obase + i]) * sc);
      *(u32x4*)(Outp + ((size_t)(b * N_ + n0 + r)) * CQK_ + obase) = pk.q;
    }
  }

  // ---- v projection: 16 output tiles, stored transposed Vt[b][c][n] ----
  #pragma unroll
  for (int ot = 0; ot < 16; ++ot) {
    const int orow = ot * 16 + r;
    v8f acc = {};
    #pragma unroll
    for (int kc = 0; kc < 8; ++kc) {
      V16U a;
      const float* wr = wv + orow * C_ + kc * 32;
      #pragma unroll
      for (int j = 0; j < 8; ++j) {
        a.v[j]     = f2bf(wr[hf * 8 + j]);
        a.v[8 + j] = f2bf(wr[16 + hf * 8 + j]);
      }
      V16U bm;
      const u32x4* bp = (const u32x4*)&Xs[w][r][kc * 32 + hf * 8];
      bm.q[0] = bp[0]; bm.q[1] = bp[2];
      acc = wmma_bf16(a.v, bm.v, acc);
    }
    const int obase = ot * 16 + hf * 8;
    #pragma unroll
    for (int i = 0; i < 8; ++i) {
      const int c = obase + i;
      Vt[((size_t)(b * C_ + c)) * N_ + n0 + r] = f2bf(acc[i] + bv[c]);
    }
  }
}

// ---------------------------------------------------------------------------
// Phase 2: flash attention.  One wave = one 16-query tile, full C=256 output
// in 128 f32 accumulator VGPRs.  Keys processed 64 at a time; softmax uses
// native exp2 (Q was pre-scaled by log2 e).  Accumulator rescale is guarded
// by a wave-uniform branch (taken ~ln(N)/64 of the time).
// ---------------------------------------------------------------------------
__global__ __launch_bounds__(128) void attn_kernel(
    const float*  __restrict__ x,
    const __bf16* __restrict__ Qh,
    const __bf16* __restrict__ Kh,
    const __bf16* __restrict__ Vt,
    float* __restrict__ out)
{
  // P staging: [16 rows][64 keys], row padded to 72 bf16 (144B = 9*16B):
  // 16B-aligned b128 reads, conflict-free (stride 36 dwords, 9 coprime 16).
  __shared__ __attribute__((aligned(16))) __bf16 Ps[4][16][72];

  const int w    = threadIdx.x >> 5;
  const int lane = threadIdx.x & 31;
  const int hf   = lane >> 4;
  const int r    = lane & 15;
  const int gt   = blockIdx.x * 4 + w;
  const int b    = gt / NT16_;
  const int q0   = (gt % NT16_) * 16;

  // resident Q A-tile: row = q0 + r, K pattern (j/8)*16 + hf*8 + j%8
  V16U qa;
  {
    const u32x4* qp = (const u32x4*)(Qh + ((size_t)(b * N_ + q0 + r)) * CQK_);
    qa.q[0] = qp[hf];
    qa.q[1] = qp[2 + hf];
  }

  v8f Oacc[16];
  const v8f vzero = {};
  #pragma unroll
  for (int t = 0; t < 16; ++t) Oacc[t] = vzero;
  float mstat[8], lstat[8];
  #pragma unroll
  for (int i = 0; i < 8; ++i) { mstat[i] = -3.0e38f; lstat[i] = 0.0f; }

  for (int kb = 0; kb < N_; kb += 64) {
    // ---- S = Q @ K^T for 64 keys (four 16-key tiles) ----
    v8f s[4];
    #pragma unroll
    for (int t = 0; t < 4; ++t) {
      V16U kbm;
      const u32x4* kp =
          (const u32x4*)(Kh + ((size_t)(b * N_ + kb + t * 16 + r)) * CQK_ + hf * 8);
      kbm.q[0] = kp[0]; kbm.q[1] = kp[2];
      s[t] = wmma_bf16(qa.v, kbm.v, vzero);
    }

    // ---- new row maxes (row m = hf*8+i lives across 16 lanes of a half) ----
    float mnew[8];
    int upd = 0;
    #pragma unroll
    for (int i = 0; i < 8; ++i) {
      float rm = fmaxf(fmaxf(s[0][i], s[1][i]), fmaxf(s[2][i], s[3][i]));
      #pragma unroll
      for (int off = 1; off < 16; off <<= 1)
        rm = fmaxf(rm, __shfl_xor(rm, off, 32));
      mnew[i] = fmaxf(mstat[i], rm);
      upd |= (mnew[i] > mstat[i]) ? 1 : 0;
    }

    // ---- wave-uniform rescale, only when some row max grew ----
    if (__any(upd)) {
      #pragma unroll
      for (int i = 0; i < 8; ++i) {
        const float scale = __builtin_amdgcn_exp2f(mstat[i] - mnew[i]);
        lstat[i] *= scale;
        mstat[i] = mnew[i];
        #pragma unroll
        for (int ct = 0; ct < 16; ++ct) Oacc[ct][i] *= scale;
      }
    }

    // ---- probabilities + row sums, stage P (D layout) into LDS ----
    #pragma unroll
    for (int i = 0; i < 8; ++i) {
      float rs = 0.0f;
      #pragma unroll
      for (int t = 0; t < 4; ++t) {
        const float p = __builtin_amdgcn_exp2f(s[t][i] - mstat[i]);
        rs += p;
        Ps[w][hf * 8 + i][t * 16 + r] = f2bf(p);
      }
      #pragma unroll
      for (int off = 1; off < 16; off <<= 1)
        rs += __shfl_xor(rs, off, 32);
      lstat[i] += rs;
    }
    lds_fence_wave();

    // ---- reload P in A layout: row = r, K = ks*32 + (j/8)*16 + hf*8 + j%8 ----
    V16U pa[2];
    {
      const __bf16* rowp = &Ps[w][r][0];
      #pragma unroll
      for (int ks = 0; ks < 2; ++ks) {
        pa[ks].q[0] = *(const u32x4*)(rowp + ks * 32 + hf * 8);
        pa[ks].q[1] = *(const u32x4*)(rowp + ks * 32 + 16 + hf * 8);
      }
    }

    // ---- O += P @ V over 16 channel tiles x 2 K-steps ----
    #pragma unroll
    for (int ct = 0; ct < 16; ++ct) {
      #pragma unroll
      for (int ks = 0; ks < 2; ++ks) {
        V16U vb;
        const u32x4* vp = (const u32x4*)(
            Vt + ((size_t)(b * C_ + ct * 16 + r)) * N_ + kb + ks * 32 + hf * 8);
        vb.q[0] = vp[0]; vb.q[1] = vp[2];
        Oacc[ct] = wmma_bf16(pa[ks].v, vb.v, Oacc[ct]);
      }
    }
  }

  // ---- epilogue: normalize, residual add, transposed contiguous stores ----
  float inv[8];
  #pragma unroll
  for (int i = 0; i < 8; ++i) inv[i] = __builtin_amdgcn_rcpf(lstat[i]);
  #pragma unroll
  for (int ct = 0; ct < 16; ++ct) {
    const int c = ct * 16 + r;
    const size_t base = ((size_t)(b * C_ + c)) * N_ + q0 + hf * 8;
    f32x4 x0 = *(const f32x4*)(x + base);
    f32x4 x1 = *(const f32x4*)(x + base + 4);
    f32x4 o0, o1;
    o0.x = Oacc[ct][0] * inv[0] + x0.x;
    o0.y = Oacc[ct][1] * inv[1] + x0.y;
    o0.z = Oacc[ct][2] * inv[2] + x0.z;
    o0.w = Oacc[ct][3] * inv[3] + x0.w;
    o1.x = Oacc[ct][4] * inv[4] + x1.x;
    o1.y = Oacc[ct][5] * inv[5] + x1.y;
    o1.z = Oacc[ct][6] * inv[6] + x1.z;
    o1.w = Oacc[ct][7] * inv[7] + x1.w;
    *(f32x4*)(out + base)     = o0;
    *(f32x4*)(out + base + 4) = o1;
  }
}

// ---------------------------------------------------------------------------
extern "C" void kernel_launch(void* const* d_in, const int* in_sizes, int n_in,
                              void* d_out, int out_size, void* d_ws, size_t ws_size,
                              hipStream_t stream) {
  const float* x  = (const float*)d_in[0];
  const float* wq = (const float*)d_in[1];
  const float* bq = (const float*)d_in[2];
  const float* wk = (const float*)d_in[3];
  const float* bk = (const float*)d_in[4];
  const float* wv = (const float*)d_in[5];
  const float* bv = (const float*)d_in[6];
  float* out = (float*)d_out;

  char* ws = (char*)d_ws;
  const size_t qk_bytes = (size_t)B_ * N_ * CQK_ * 2;   // 1 MB each
  __bf16* Qh = (__bf16*)(ws);
  __bf16* Kh = (__bf16*)(ws + qk_bytes);
  __bf16* Vt = (__bf16*)(ws + 2 * qk_bytes);            // 8 MB

  proj_kernel<<<dim3((B_ * NT16_) / 4), dim3(128), 0, stream>>>(
      x, wq, bq, wk, bk, wv, bv, Qh, Kh, Vt);
  attn_kernel<<<dim3((B_ * NT16_) / 4), dim3(128), 0, stream>>>(
      x, Qh, Kh, Vt, out);
}
